// GNNDrugInteractionModel_79766132621709
// MI455X (gfx1250) — compile-verified, matrix operand order checked by reading, not run on hardware
//
#include <hip/hip_runtime.h>
#include <hip/hip_fp16.h>

typedef __attribute__((ext_vector_type(16))) _Float16 v16h;
typedef __attribute__((ext_vector_type(8)))  float    v8f;

#define N_NODES  100000
#define N_EDGES  3200000
#define N_GRAPHS 2048
#define F_IN     128
#define F_HID    64

// ---------------------------------------------------------------- utilities

__global__ void fill_kernel(float* __restrict__ p, float v, int n) {
    int i = blockIdx.x * blockDim.x + threadIdx.x;
    if (i < n) p[i] = v;
}

// degree: deg[col[e]] += 1  (deg pre-filled with 1.0 for self loop)
__global__ void deg_accum_kernel(const int* __restrict__ col, float* __restrict__ deg, int E) {
    int i = blockIdx.x * blockDim.x + threadIdx.x;
    if (i < E) unsafeAtomicAdd(&deg[col[i]], 1.0f);
}

__global__ void dinv_kernel(float* __restrict__ deg, int n) {
    int i = blockIdx.x * blockDim.x + threadIdx.x;
    if (i < n) deg[i] = rsqrtf(deg[i]);
}

// ---------------------------------------------------------------- WMMA GEMM
// Out[M x 64] = X[M x K] @ W[K x 64]; one wave per 16x16 output tile.
// A/B operand layouts per CDNA5 ISA 7.12.2 (16-bit, 16x32 per K-step):
//   lane L (L<16): row/col = L, holds K = kk+{0..7} and kk+16+{0..7}
//   lane L (>=16): row/col = L-16, holds K = kk+8+{0..7} and kk+24+{0..7}
template <int K>
__global__ void wmma_gemm16_kernel(const float* __restrict__ X,
                                   const float* __restrict__ W,
                                   float* __restrict__ Out) {
    const int lane = threadIdx.x & 31;
    const int m0   = blockIdx.x * 16;
    const int n0   = blockIdx.y * 16;
    const int mr   = lane & 15;
    const int kb   = (lane >> 4) * 8;        // 0 or 8
    const int col  = n0 + mr;

    const float* xrow = X + (size_t)(m0 + mr) * K;

    v8f c = {};
#pragma unroll
    for (int kk = 0; kk < K; kk += 32) {
        v16h a, b;
#pragma unroll
        for (int e = 0; e < 8; ++e) {
            a[e]     = (_Float16)xrow[kk + kb + e];
            a[8 + e] = (_Float16)xrow[kk + 16 + kb + e];
            b[e]     = (_Float16)W[(size_t)(kk + kb + e) * F_HID + col];
            b[8 + e] = (_Float16)W[(size_t)(kk + 16 + kb + e) * F_HID + col];
        }
        c = __builtin_amdgcn_wmma_f32_16x16x32_f16(false, a, false, b,
                                                   (short)0, c, false, false);
    }

#pragma unroll
    for (int r = 0; r < 8; ++r) {
        int m = m0 + ((lane < 16) ? r : (8 + r));
        Out[(size_t)m * F_HID + n0 + mr] = c[r];
    }
}

// ------------------------------------------------------- edge message pass
// One wave per edge: agg[col] += h[row] * dinv[row]*dinv[col]; 2 floats/lane.
__global__ void edge_msg_kernel(const int* __restrict__ row,
                                const int* __restrict__ col,
                                const float* __restrict__ dinv,
                                const float* __restrict__ H,
                                float* __restrict__ Agg) {
    const int warp = threadIdx.x >> 5;
    const int lane = threadIdx.x & 31;
    const long e = (long)blockIdx.x * (blockDim.x >> 5) + warp;
    if (e >= N_EDGES) return;
    const int r = row[e];
    const int c = col[e];
    const float norm = dinv[r] * dinv[c];
    const float* hs = H   + (size_t)r * F_HID;
    float*       ag = Agg + (size_t)c * F_HID;
    unsafeAtomicAdd(&ag[lane],      hs[lane]      * norm);
    unsafeAtomicAdd(&ag[lane + 32], hs[lane + 32] * norm);
}

// Agg = relu(Agg + H*dinv^2 + bias)
__global__ void fuse_selfloop_bias_relu_kernel(float* __restrict__ Agg,
                                               const float* __restrict__ H,
                                               const float* __restrict__ dinv,
                                               const float* __restrict__ bias) {
    long i = (long)blockIdx.x * blockDim.x + threadIdx.x;
    if (i >= (long)N_NODES * F_HID) return;
    const int node = (int)(i >> 6);
    const int f    = (int)(i & 63);
    const float d = dinv[node];
    float v = Agg[i] + H[i] * d * d + bias[f];
    Agg[i] = v > 0.0f ? v : 0.0f;
}

// -------------------------------------------------------------- mean pool
__global__ void pool_accum_kernel(const float* __restrict__ H,
                                  const int* __restrict__ batch,
                                  float* __restrict__ psum,
                                  float* __restrict__ cnt) {
    const int warp = threadIdx.x >> 5;
    const int lane = threadIdx.x & 31;
    const long node = (long)blockIdx.x * (blockDim.x >> 5) + warp;
    if (node >= N_NODES) return;
    const int g = batch[node];
    const float* hs = H + (size_t)node * F_HID;
    float* ps = psum + (size_t)g * F_HID;
    unsafeAtomicAdd(&ps[lane],      hs[lane]);
    unsafeAtomicAdd(&ps[lane + 32], hs[lane + 32]);
    if (lane == 0) unsafeAtomicAdd(&cnt[g], 1.0f);
}

// ------------------------------------------------------------- MLP head
// One wave per graph: emb = psum/cnt; h = relu(emb@fcW1+fcb1) (lane j owns
// output j of 32); out = sigmoid(h@fcW2+fcb2) via wave32 shuffle reduce.
__global__ void head_kernel(const float* __restrict__ psum,
                            const float* __restrict__ cnt,
                            const float* __restrict__ fcW1,
                            const float* __restrict__ fcb1,
                            const float* __restrict__ fcW2,
                            const float* __restrict__ fcb2,
                            float* __restrict__ out) {
    const int warp = threadIdx.x >> 5;
    const int lane = threadIdx.x & 31;
    const int g = blockIdx.x * (blockDim.x >> 5) + warp;
    if (g >= N_GRAPHS) return;

    float c = cnt[g];
    c = c > 1.0f ? c : 1.0f;
    const float inv = 1.0f / c;

    const float* emb = psum + (size_t)g * F_HID;
    float acc = fcb1[lane];
#pragma unroll
    for (int k = 0; k < F_HID; ++k)
        acc += emb[k] * inv * fcW1[k * 32 + lane];
    float h = acc > 0.0f ? acc : 0.0f;

    float p = h * fcW2[lane];
#pragma unroll
    for (int off = 16; off > 0; off >>= 1)
        p += __shfl_xor(p, off, 32);

    if (lane == 0) {
        float z = p + fcb2[0];
        out[g] = 1.0f / (1.0f + __expf(-z));
    }
}

// ---------------------------------------------------------------- launcher

extern "C" void kernel_launch(void* const* d_in, const int* in_sizes, int n_in,
                              void* d_out, int out_size, void* d_ws, size_t ws_size,
                              hipStream_t stream) {
    const float* x     = (const float*)d_in[0];
    const int*   ei    = (const int*)d_in[1];
    const int*   batch = (const int*)d_in[2];
    const float* W1    = (const float*)d_in[3];
    const float* b1    = (const float*)d_in[4];
    const float* W2    = (const float*)d_in[5];
    const float* b2    = (const float*)d_in[6];
    const float* fcW1  = (const float*)d_in[7];
    const float* fcb1  = (const float*)d_in[8];
    const float* fcW2  = (const float*)d_in[9];
    const float* fcb2  = (const float*)d_in[10];
    float* out = (float*)d_out;

    const int* rowi = ei;            // edge_index[0]
    const int* coli = ei + N_EDGES;  // edge_index[1]

    float* f    = (float*)d_ws;
    float* dinv = f;                                    // N_NODES
    float* A    = dinv + N_NODES;                       // N_NODES*F_HID (h)
    float* B    = A + (size_t)N_NODES * F_HID;          // N_NODES*F_HID (agg)
    float* psum = B + (size_t)N_NODES * F_HID;          // N_GRAPHS*F_HID
    float* cnt  = psum + (size_t)N_GRAPHS * F_HID;      // N_GRAPHS (contiguous)

    const int NF = N_NODES * F_HID;

    // --- degrees (with self loop) -> dinv
    fill_kernel<<<(N_NODES + 255) / 256, 256, 0, stream>>>(dinv, 1.0f, N_NODES);
    deg_accum_kernel<<<(N_EDGES + 255) / 256, 256, 0, stream>>>(coli, dinv, N_EDGES);
    dinv_kernel<<<(N_NODES + 255) / 256, 256, 0, stream>>>(dinv, N_NODES);

    dim3 ggrid(N_NODES / 16, F_HID / 16);   // 6250 x 4, no tails

    // --- layer 1: A = x @ W1 ; B = scatter(msgs) ; B = relu(B + A*dinv^2 + b1)
    wmma_gemm16_kernel<F_IN><<<ggrid, 32, 0, stream>>>(x, W1, A);
    fill_kernel<<<(NF + 255) / 256, 256, 0, stream>>>(B, 0.0f, NF);
    edge_msg_kernel<<<(N_EDGES + 7) / 8, 256, 0, stream>>>(rowi, coli, dinv, A, B);
    fuse_selfloop_bias_relu_kernel<<<(NF + 255) / 256, 256, 0, stream>>>(B, A, dinv, b1);

    // --- layer 2: A = B @ W2 ; reuse B as agg2
    wmma_gemm16_kernel<F_HID><<<ggrid, 32, 0, stream>>>(B, W2, A);
    fill_kernel<<<(NF + 255) / 256, 256, 0, stream>>>(B, 0.0f, NF);
    edge_msg_kernel<<<(N_EDGES + 7) / 8, 256, 0, stream>>>(rowi, coli, dinv, A, B);
    fuse_selfloop_bias_relu_kernel<<<(NF + 255) / 256, 256, 0, stream>>>(B, A, dinv, b2);

    // --- global mean pool
    fill_kernel<<<(N_GRAPHS * (F_HID + 1) + 255) / 256, 256, 0, stream>>>(
        psum, 0.0f, N_GRAPHS * (F_HID + 1));
    pool_accum_kernel<<<(N_NODES + 7) / 8, 256, 0, stream>>>(B, batch, psum, cnt);

    // --- MLP head
    head_kernel<<<(N_GRAPHS + 7) / 8, 256, 0, stream>>>(psum, cnt, fcW1, fcb1,
                                                        fcW2, fcb2, out);
}
